// SR_Matcher_22333829939702
// MI455X (gfx1250) — matrix operand order repaired and checked by reading, not compiled, not run on hardware
//
#include <hip/hip_runtime.h>
#include <hip/hip_bf16.h>

typedef __attribute__((ext_vector_type(16))) _Float16 v16h;
typedef __attribute__((ext_vector_type(8)))  _Float16 v8h;
typedef __attribute__((ext_vector_type(8)))  float    v8f;

#define B_DIM 32
#define S_LEN 1024
#define V_DIM 128
#define EMB_DIM 768
#define H1_DIM 300
#define D_DIM 200
#define M_ROWS (B_DIM * S_LEN)   // 32768

// padded dims
#define H1P 320     // H1 padded to mult of 64
#define DP  224     // D padded to mult of 32 (ld of Q/Y/memh)
#define DNP 256     // padded N extent covered by weight buffers (mult of 64)

__device__ inline v8f wmma_f16(v16h a, v16h b, v8f c) {
    return __builtin_amdgcn_wmma_f32_16x16x32_f16(
        false, a, false, b, (short)0, c, false, false);
}

__device__ inline v16h cat16(v8h lo, v8h hi) {
    return __builtin_shufflevector(lo, hi, 0,1,2,3,4,5,6,7,8,9,10,11,12,13,14,15);
}

// ---- fragment loaders: NO bounds checks, 16B-aligned vector loads ----------
// A (16x32 f16): lane m=lane&15 ; e0..7 -> k=8*(lane>>4)+e ; e8..15 -> k=16+8*(lane>>4)+e
__device__ inline v16h load_a(const _Float16* __restrict__ A, int ld,
                              int m0, int k0, int lane) {
    int m  = m0 + (lane & 15);
    int kb = (lane >> 4) << 3;
    const _Float16* p = A + (long long)m * ld + (k0 + kb);
    v8h lo = *(const v8h*)p;
    v8h hi = *(const v8h*)(p + 16);
    return cat16(lo, hi);
}
__device__ inline v16h load_a(const float* __restrict__ A, int ld,
                              int m0, int k0, int lane) {
    int m  = m0 + (lane & 15);
    int kb = (lane >> 4) << 3;
    const float* p = A + (long long)m * ld + (k0 + kb);
    v16h a;
#pragma unroll
    for (int e = 0; e < 8; ++e) {
        a[e]     = (_Float16)p[e];
        a[e + 8] = (_Float16)p[16 + e];
    }
    return a;
}
// B (32x16 f16) from transposed source Bt[n][k]: lane n=lane&15 ; e0..7 -> k=8*(lane>>4)+e ; e8..15 -> k=16+8*(lane>>4)+e
__device__ inline v16h load_bt(const _Float16* __restrict__ Bt, int ldk,
                               int k0, int n0, int lane) {
    int n  = n0 + (lane & 15);
    int kb = (lane >> 4) << 3;
    const _Float16* p = Bt + (long long)n * ldk + (k0 + kb);
    v8h lo = *(const v8h*)p;
    v8h hi = *(const v8h*)(p + 16);
    return cat16(lo, hi);
}

// ---------------------------------------------------------------------------
// GEMM: C[M_ROWS,N] = act(A[M_ROWS,K] @ Bt^T + bias). lda==ldk==K (padded).
// 8 waves/block; each wave: 16(M) x 64(N), 4 accumulators, branch-free k-loop.
// grid = (ceil(N/64), M_ROWS/128), block = 256.
// ---------------------------------------------------------------------------
template <typename AT, typename OT>
__global__ __launch_bounds__(256) void gemm_wmma(
    const AT* __restrict__ A, const _Float16* __restrict__ Bt,
    const float* __restrict__ bias, OT* __restrict__ C,
    int N, int K, int ldc, int do_relu) {

    int lane  = threadIdx.x & 31;
    int wave  = threadIdx.x >> 5;
    int mbase = blockIdx.y * 128 + wave * 16;
    int nbase = blockIdx.x * 64;

    v8f acc0 = {}, acc1 = {}, acc2 = {}, acc3 = {};

    for (int k0 = 0; k0 < K; k0 += 32) {
        v16h a = load_a(A, K, mbase, k0, lane);
        if (k0 + 32 < K)   // uniform branch: prefetch next A tile
            __builtin_prefetch(A + (long long)(mbase + (lane & 15)) * K + k0 + 32, 0, 1);
        acc0 = wmma_f16(a, load_bt(Bt, K, k0, nbase +  0, lane), acc0);
        acc1 = wmma_f16(a, load_bt(Bt, K, k0, nbase + 16, lane), acc1);
        acc2 = wmma_f16(a, load_bt(Bt, K, k0, nbase + 32, lane), acc2);
        acc3 = wmma_f16(a, load_bt(Bt, K, k0, nbase + 48, lane), acc3);
    }

    int l16   = lane & 15;
    int hf    = lane >> 4;
    int mrow0 = mbase + (hf << 3);
    v8f accs[4] = {acc0, acc1, acc2, acc3};
#pragma unroll
    for (int t = 0; t < 4; ++t) {
        int n = nbase + t * 16 + l16;
        if (n >= N) continue;                    // epilogue-only guard
        float bv = bias ? bias[n] : 0.0f;
#pragma unroll
        for (int r = 0; r < 8; ++r) {
            float v = accs[t][r] + bv;
            if (do_relu) v = fmaxf(v, 0.0f);
            C[(long long)(mrow0 + r) * ldc + n] = (OT)v;
        }
    }
}

// ---------------------------------------------------------------------------
// Fused flash attention: per batch, Omid = softmax(Q @ Y^T) @ SRL.  All f16
// operands, f32 accum/softmax. block = 256 (8 waves, 16 q-rows each);
// grid = (S/128, B). SRL value tile staged transposed in LDS per key-step.
// ---------------------------------------------------------------------------
__global__ __launch_bounds__(256) void attn_flash(
    const _Float16* __restrict__ Q,    // [B*S, DP]
    const _Float16* __restrict__ Y,    // [B*S, DP]
    const _Float16* __restrict__ SRL,  // [B*S, V]
    _Float16* __restrict__ O) {        // [B*S, V]

    __shared__ _Float16 Pb[8][16][32];   // per-wave P transpose (C-layout -> A-layout)
    __shared__ _Float16 Vt[V_DIM][40];   // value tile transposed: Vt[v][key], stride 40 (16B-aligned rows)

    int lane  = threadIdx.x & 31;
    int wave  = threadIdx.x >> 5;
    int hf    = lane >> 4;
    int l16   = lane & 15;
    int b     = blockIdx.y;
    int qbase = blockIdx.x * 128 + wave * 16;

    const _Float16* Qb = Q   + (long long)b * S_LEN * DP;
    const _Float16* Yb = Y   + (long long)b * S_LEN * DP;
    const _Float16* Vb = SRL + (long long)b * S_LEN * V_DIM;

    v8f o[8];
#pragma unroll
    for (int t = 0; t < 8; ++t) o[t] = (v8f){};
    float mrun[8], lrun[8];
#pragma unroll
    for (int r = 0; r < 8; ++r) { mrun[r] = -1e30f; lrun[r] = 0.0f; }

    for (int j0 = 0; j0 < S_LEN; j0 += 32) {
        // ---- scores S[16q x 32k] = Q_tile @ Y_tile^T (branch-free, b128) ----
        v8f s0 = {}, s1 = {};
        for (int k0 = 0; k0 < DP; k0 += 32) {
            v16h aq = load_a(Qb, DP, qbase, k0, lane);
            s0 = wmma_f16(aq, load_bt(Yb, DP, k0, j0,      lane), s0);
            s1 = wmma_f16(aq, load_bt(Yb, DP, k0, j0 + 16, lane), s1);
        }

        // ---- online softmax (row reductions across 16-lane halves) ----
        float alpha[8];
        v8f p0, p1;
#pragma unroll
        for (int r = 0; r < 8; ++r) {
            float t = fmaxf(s0[r], s1[r]);
#pragma unroll
            for (int off = 1; off < 16; off <<= 1)
                t = fmaxf(t, __shfl_xor(t, off, 16));
            float mn = fmaxf(mrun[r], t);
            alpha[r] = __expf(mrun[r] - mn);
            float e0 = __expf(s0[r] - mn);
            float e1 = __expf(s1[r] - mn);
            p0[r] = e0; p1[r] = e1;
            float ps = e0 + e1;
#pragma unroll
            for (int off = 1; off < 16; off <<= 1)
                ps += __shfl_xor(ps, off, 16);
            lrun[r] = lrun[r] * alpha[r] + ps;
            mrun[r] = mn;
        }
#pragma unroll
        for (int t = 0; t < 8; ++t)
#pragma unroll
            for (int r = 0; r < 8; ++r) o[t][r] *= alpha[r];

        // ---- LDS: write P (transpose) + stage value tile, one sync pair ----
        __syncthreads();   // protect previous iteration's LDS reads
#pragma unroll
        for (int r = 0; r < 8; ++r) {
            int m = r + 8 * hf;
            Pb[wave][m][l16]      = (_Float16)p0[r];
            Pb[wave][m][16 + l16] = (_Float16)p1[r];
        }
        {   // cooperative staging: 32 keys x 128 v, transposed into Vt[v][key]
            int key = threadIdx.x >> 3;          // 0..31
            int v0  = (threadIdx.x & 7) * 16;    // 0..112
            const _Float16* src = Vb + (long long)(j0 + key) * V_DIM + v0;
            v8h d0 = *(const v8h*)src;
            v8h d1 = *(const v8h*)(src + 8);
#pragma unroll
            for (int e = 0; e < 8; ++e) {
                Vt[v0 + e][key]     = d0[e];
                Vt[v0 + 8 + e][key] = d1[e];
            }
        }
        __syncthreads();

        // P in A-layout: two aligned 16B LDS reads
        v16h pa;
        {
            int kb = hf * 8;
            v8h lo = *(const v8h*)&Pb[wave][l16][kb];
            v8h hi = *(const v8h*)&Pb[wave][l16][16 + kb];
            pa = cat16(lo, hi);
        }

        // ---- O[16 x 128] += P[16 x 32] @ V[32 x 128] (B-frags from LDS) ----
#pragma unroll
        for (int t = 0; t < 8; ++t) {
            int v  = t * 16 + l16;
            int kb = hf << 3;
            v8h lo = *(const v8h*)&Vt[v][kb];
            v8h hi = *(const v8h*)&Vt[v][kb + 16];
            o[t] = wmma_f16(pa, cat16(lo, hi), o[t]);
        }
    }

    // ---- normalize + store f16 ----
    _Float16* Ob = O + (long long)b * S_LEN * V_DIM;
#pragma unroll
    for (int t = 0; t < 8; ++t) {
        int v = t * 16 + l16;
#pragma unroll
        for (int r = 0; r < 8; ++r) {
            int m = qbase + r + 8 * hf;
            Ob[(long long)m * V_DIM + v] = (_Float16)(o[t][r] / lrun[r]);
        }
    }
}

// ---------------------------------------------------------------------------
// Prep kernels: pad/transpose/convert into f16 workspace buffers.
// ---------------------------------------------------------------------------
// out[n*Kp + k] = (k<K && n<N) ? in[k*N + n] : 0   (weight transpose+pad)
__global__ void transpose_pad_f16(const float* __restrict__ in, _Float16* __restrict__ out,
                                  int K, int N, int Kp, int Np) {
    int idx = blockIdx.x * blockDim.x + threadIdx.x;
    if (idx >= Kp * Np) return;
    int k = idx % Kp, n = idx / Kp;
    float v = (k < K && n < N) ? in[(long long)k * N + n] : 0.0f;
    out[idx] = (_Float16)v;
}
// out[r*cout + c] = (c<cin) ? in[r*cin + c] : 0    (row-wise convert+pad)
__global__ void convert_rows_f16(const float* __restrict__ in, _Float16* __restrict__ out,
                                 int cin, int cout, long long total) {
    long long idx = (long long)blockIdx.x * blockDim.x + threadIdx.x;
    if (idx >= total) return;
    long long r = idx / cout;
    int c = (int)(idx - r * cout);
    out[idx] = (c < cin) ? (_Float16)in[r * cin + c] : (_Float16)0.0f;
}
__global__ void pad_bias_f32(const float* __restrict__ in, float* __restrict__ out, int N, int Np) {
    int n = blockIdx.x * blockDim.x + threadIdx.x;
    if (n < Np) out[n] = (n < N) ? in[n] : 0.0f;
}

// ---------------------------------------------------------------------------
extern "C" void kernel_launch(void* const* d_in, const int* in_sizes, int n_in,
                              void* d_out, int out_size, void* d_ws, size_t ws_size,
                              hipStream_t stream) {
    const float* mem = (const float*)d_in[0];   // [B,S,D]
    const float* SRL = (const float*)d_in[1];   // [B,S,V]
    const float* emb = (const float*)d_in[2];   // [B,S,EMB]
    const float* W1  = (const float*)d_in[4];   // [EMB,H1]
    const float* b1  = (const float*)d_in[5];   // [H1]
    const float* W2  = (const float*)d_in[6];   // [H1,D]
    const float* b2  = (const float*)d_in[7];   // [D]
    const float* Mx  = (const float*)d_in[8];   // [D,D]
    const float* Wp  = (const float*)d_in[9];   // [V,V]
    const float* bp  = (const float*)d_in[10];  // [V]
    float*       out = (float*)d_out;           // [B*S, V] f32

    char* ws = (char*)d_ws;
    size_t off = 0;
    auto alloc = [&](size_t bytes) { void* p = ws + off; off += (bytes + 255) & ~(size_t)255; return p; };

    _Float16* H    = (_Float16*)alloc((size_t)M_ROWS * H1P  * 2);
    _Float16* Qm   = (_Float16*)alloc((size_t)M_ROWS * DP   * 2);
    _Float16* Ym   = (_Float16*)alloc((size_t)M_ROWS * DP   * 2);
    _Float16* memh = (_Float16*)alloc((size_t)M_ROWS * DP   * 2);
    _Float16* SRLh = (_Float16*)alloc((size_t)M_ROWS * V_DIM * 2);
    _Float16* Omid = (_Float16*)alloc((size_t)M_ROWS * V_DIM * 2);
    _Float16* W1t  = (_Float16*)alloc((size_t)H1P * EMB_DIM * 2);
    _Float16* W2t  = (_Float16*)alloc((size_t)DNP * H1P * 2);
    _Float16* Mxt  = (_Float16*)alloc((size_t)DNP * DP * 2);
    _Float16* Wph  = (_Float16*)alloc((size_t)V_DIM * V_DIM * 2);
    float*    b1p  = (float*)alloc(H1P * 4);
    float*    b2p  = (float*)alloc(DNP * 4);

    dim3 blk(256);

    // ---- prep: padded/transposed f16 weights, padded biases, f16 inputs ----
    transpose_pad_f16<<<(H1P * EMB_DIM + 255) / 256, blk, 0, stream>>>(W1, W1t, EMB_DIM, H1_DIM, EMB_DIM, H1P);
    transpose_pad_f16<<<(DNP * H1P + 255) / 256, blk, 0, stream>>>(W2, W2t, H1_DIM, D_DIM, H1P, DNP);
    transpose_pad_f16<<<(DNP * DP + 255) / 256, blk, 0, stream>>>(Mx, Mxt, D_DIM, D_DIM, DP, DNP);
    {
        long long tot = (long long)V_DIM * V_DIM;      // Wp used untransposed: Bt[j][v] = Wp[j][v]
        convert_rows_f16<<<(unsigned)((tot + 255) / 256), blk, 0, stream>>>(Wp, Wph, V_DIM, V_DIM, tot);
    }
    pad_bias_f32<<<(H1P + 255) / 256, blk, 0, stream>>>(b1, b1p, H1_DIM, H1P);
    pad_bias_f32<<<(DNP + 255) / 256, blk, 0, stream>>>(b2, b2p, D_DIM, DNP);
    {
        long long tot = (long long)M_ROWS * DP;
        convert_rows_f16<<<(unsigned)((tot + 255) / 256), blk, 0, stream>>>(mem, memh, D_DIM, DP, tot);
    }
    {
        long long tot = (long long)M_ROWS * V_DIM;
        convert_rows_f16<<<(unsigned)((tot + 255) / 256), blk, 0, stream>>>(SRL, SRLh, V_DIM, V_DIM, tot);
    }

    // ---- pipeline ----
    // 1) H = relu(emb @ W1 + b1) : f32 A on-the-fly, f16 out [32768 x 320]
    gemm_wmma<float, _Float16><<<dim3(H1P / 64, M_ROWS / 128), blk, 0, stream>>>(
        emb, W1t, b1p, H, H1P, EMB_DIM, H1P, 1);
    // 2) Q = relu(H @ W2 + b2) : [32768 x 224] (cols 200.. zero via padded weights)
    gemm_wmma<_Float16, _Float16><<<dim3((DP + 63) / 64, M_ROWS / 128), blk, 0, stream>>>(
        H, W2t, b2p, Qm, DP, H1P, DP, 1);
    // 3) Y = mem @ matrix : [32768 x 224]
    gemm_wmma<_Float16, _Float16><<<dim3((DP + 63) / 64, M_ROWS / 128), blk, 0, stream>>>(
        memh, Mxt, nullptr, Ym, DP, DP, DP, 0);
    // 4) Omid = softmax(Q @ Y^T) @ SRL (fused flash, per batch)
    attn_flash<<<dim3(S_LEN / 128, B_DIM), blk, 0, stream>>>(Qm, Ym, SRLh, Omid);
    // 5) out = Omid @ Wp^T + bp : f32 out [32768 x 128]
    gemm_wmma<_Float16, float><<<dim3(V_DIM / 64, M_ROWS / 128), blk, 0, stream>>>(
        Omid, Wph, bp, out, V_DIM, V_DIM, V_DIM, 0);
}
